// TransformerBlock_17746804867396
// MI455X (gfx1250) — matrix operand order repaired and checked by reading, not compiled, not run on hardware
//
#include <hip/hip_runtime.h>
#include <cstdint>
#include <cstddef>

#define D_MODEL 1024
#define N_HEADS 16
#define HEAD_DIM 64
#define D_FF 4096
#define BATCH 2
#define SEQ 2048
#define ROWS (BATCH*SEQ)   /* 4096 */

typedef __attribute__((ext_vector_type(16))) __bf16 v16bf;
typedef __attribute__((ext_vector_type(8)))  float v8f;
typedef __attribute__((ext_vector_type(4)))  unsigned int v4u;
typedef __attribute__((ext_vector_type(8)))  int v8i;
typedef __attribute__((ext_vector_type(4)))  int v4i;

#if defined(__gfx1250__) && __has_builtin(__builtin_amdgcn_tensor_load_to_lds)
#define USE_TDM 1
#else
#define USE_TDM 0
#endif

union FragB16 {
  unsigned int   u[8];
  unsigned short s[16];
  v16bf          v;
};

__device__ __forceinline__ unsigned short f2bf(float f) {
  union { float f; unsigned int u; } c; c.f = f;
  unsigned int u = c.u;
  unsigned int r = u + 0x7FFFu + ((u >> 16) & 1u);   // round-to-nearest-even
  return (unsigned short)(r >> 16);
}
__device__ __forceinline__ float bf2f(unsigned short h) {
  union { unsigned int u; float f; } c; c.u = ((unsigned int)h) << 16;
  return c.f;
}
__device__ __forceinline__ v8f v8f_zero() {
  v8f z;
#pragma unroll
  for (int i = 0; i < 8; ++i) z[i] = 0.0f;
  return z;
}

#if USE_TDM
// Tensor Data Mover: DMA a 2-D bf16 tile (tile1 rows x tile0 elems) from
// global (row stride stride0 elems) into LDS at byte offset lds_addr, adding
// (pada+1) DWORDs of padding every 2<<padi DWORDs (-> padded LDS row stride).
// D# layout per CDNA5 ISA; data_size=1 (2 bytes).
// 6-arg builtin form (amdgpu-toolchain / clang-23): groups 0..3 + spare + cpol.
__device__ __forceinline__ void tdm_load_2d(unsigned lds_addr, const void* gptr,
    unsigned dim0, unsigned dim1, unsigned stride0,
    unsigned tile0, unsigned tile1, unsigned padi, unsigned pada)
{
  unsigned long long ga = (unsigned long long)(uintptr_t)gptr;
  v4u g0; v8i g1; v4i gz4; v8i gz8;
  g0[0] = 1u;                                  // count=1, user descriptor
  g0[1] = lds_addr;                            // LDS byte address
  g0[2] = (unsigned)ga;                        // global_addr[31:0]
  g0[3] = (unsigned)(ga >> 32) | 0x80000000u;  // global_addr[56:32] | type=2
  g1[0] = (int)((1u << 16) | (1u << 20) | (padi << 22) | (pada << 25));
  g1[1] = (int)((dim0 & 0xFFFFu) << 16);                    // tensor_dim0 lo
  g1[2] = (int)((dim0 >> 16) | ((dim1 & 0xFFFFu) << 16));   // dim0 hi | dim1 lo
  g1[3] = (int)((dim1 >> 16) | (tile0 << 16));              // dim1 hi | tile_dim0
  g1[4] = (int)tile1;                                       // tile_dim1 (tile_dim2=0)
  g1[5] = (int)stride0;                                     // tensor_dim0_stride lo
  g1[6] = 0; g1[7] = 0;
  gz4[0] = gz4[1] = gz4[2] = gz4[3] = 0;
#pragma unroll
  for (int i = 0; i < 8; ++i) gz8[i] = 0;
  __builtin_amdgcn_tensor_load_to_lds(g0, g1, gz4, gz4, gz8, 0);
}
#endif

// ---------------------------------------------------------------------------
// RMSNorm: one block per row of 1024, writes bf16
// ---------------------------------------------------------------------------
__global__ __launch_bounds__(256) void rmsnorm_kernel(
    const float* __restrict__ x, const float* __restrict__ w,
    unsigned short* __restrict__ out)
{
  __shared__ float red[8];
  const int row = blockIdx.x, tid = threadIdx.x;
  const float* xr = x + (size_t)row * D_MODEL;
  float v[4]; float ss = 0.0f;
#pragma unroll
  for (int i = 0; i < 4; ++i) { v[i] = xr[tid + i * 256]; ss += v[i] * v[i]; }
#pragma unroll
  for (int off = 16; off >= 1; off >>= 1) ss += __shfl_xor(ss, off, 32);
  if ((tid & 31) == 0) red[tid >> 5] = ss;
  __syncthreads();
  float tot = 0.0f;
#pragma unroll
  for (int i = 0; i < 8; ++i) tot += red[i];
  float inv = 1.0f / (sqrtf(tot * (1.0f / D_MODEL)) + 1e-6f);
#pragma unroll
  for (int i = 0; i < 4; ++i)
    out[(size_t)row * D_MODEL + tid + i * 256] = f2bf(v[i] * inv * w[tid + i * 256]);
}

// ---------------------------------------------------------------------------
// Tiled WMMA GEMM:  C[M,N] = A[M,K](bf16) * W[N,K](f32, converted)^T
//   block: 256 threads = 8 waves, tile M=128 (16 rows/wave), N=64, K-step 64
//   A tile staged by TDM (async DMA with LDS padding); W converted in-flight.
// ---------------------------------------------------------------------------
template<bool RESID>
__global__ __launch_bounds__(256) void gemm_bf16(
    const unsigned short* __restrict__ A,
    const float* __restrict__ W,
    const float* __restrict__ resid,
    float* __restrict__ outF,
    unsigned short* __restrict__ outB,
    int M, int N, int K)
{
  __shared__ unsigned short As[128][72];   // 144B rows (16B aligned)
  __shared__ unsigned short Ws[64][72];
  const int tid = threadIdx.x;
  const int lane = tid & 31, wid = tid >> 5;
  const int laneHi = lane >> 4, lane16 = lane & 15;
  const int Mbase = blockIdx.y * 128;
  const int Nbase = blockIdx.x * 64;
  const int wrow = wid * 16;

  v8f acc[4];
#pragma unroll
  for (int t = 0; t < 4; ++t) acc[t] = v8f_zero();

#if USE_TDM
  const unsigned ldsA = (unsigned)(uintptr_t)&As[0][0];
#else
  const int arow = tid >> 1, ahalf = tid & 1;        // A tile: 32 bf16 / thread
#endif
  const int wrowL = tid >> 2, wseg = tid & 3;        // W tile: 16 f32 / thread

  for (int kb = 0; kb < K; kb += 64) {
    __syncthreads();
#if USE_TDM
    if (tid < 32)  // wave 0 launches the DMA; EXEC ignored, one op per wave
      tdm_load_2d(ldsA, A + (size_t)Mbase * K + kb,
                  (unsigned)K, (unsigned)M, (unsigned)K,
                  64u, 128u, 4u, 3u);   // 32-DW rows + 4-DW pad -> 144B stride
#else
    {
      const unsigned short* g = A + (size_t)(Mbase + arow) * K + kb + ahalf * 32;
      const uint4* g4 = (const uint4*)g;
      uint4* l4 = (uint4*)&As[arow][ahalf * 32];
      l4[0] = g4[0]; l4[1] = g4[1]; l4[2] = g4[2]; l4[3] = g4[3];
    }
#endif
    { // W tile 64x64 f32 -> bf16 -> LDS
      const float* g = W + (size_t)(Nbase + wrowL) * K + kb + wseg * 16;
      unsigned int* l = (unsigned int*)&Ws[wrowL][wseg * 16];
#pragma unroll
      for (int q = 0; q < 4; ++q) {
        float4 f = ((const float4*)g)[q];
        l[2 * q + 0] = (unsigned)f2bf(f.x) | ((unsigned)f2bf(f.y) << 16);
        l[2 * q + 1] = (unsigned)f2bf(f.z) | ((unsigned)f2bf(f.w) << 16);
      }
      if (kb + 64 < K)
        __builtin_prefetch(g + 64, 0, 1);   // next K-step of W
    }
#if USE_TDM
    if (tid < 32) __builtin_amdgcn_s_wait_tensorcnt(0);
#endif
    __syncthreads();
#pragma unroll
    for (int c = 0; c < 2; ++c) {
      FragB16 a;
#pragma unroll
      for (int j = 0; j < 8; ++j) { // A layout: K = (j/4)*16 + hi*8 + (j%4)*2
        int k = c * 32 + ((j >> 2) << 4) + (laneHi << 3) + ((j & 3) << 1);
        a.u[j] = *(const unsigned int*)&As[wrow + lane16][k];
      }
#pragma unroll
      for (int nt = 0; nt < 4; ++nt) {
        FragB16 b;
#pragma unroll
        for (int j = 0; j < 8; ++j) { // B layout: K = (j/4)*16 + hi*8 + (j%4)*2
          int k = c * 32 + ((j >> 2) << 4) + (laneHi << 3) + ((j & 3) << 1);
          b.u[j] = *(const unsigned int*)&Ws[nt * 16 + lane16][k];
        }
        acc[nt] = __builtin_amdgcn_wmma_f32_16x16x32_bf16(
            false, a.v, false, b.v, (short)0, acc[nt], false, false);
      }
    }
  }
#pragma unroll
  for (int nt = 0; nt < 4; ++nt)
#pragma unroll
    for (int r = 0; r < 8; ++r) {
      int row = Mbase + wrow + laneHi * 8 + r;
      int col = Nbase + nt * 16 + lane16;
      size_t idx = (size_t)row * N + col;
      if (RESID) outF[idx] = resid[idx] + acc[nt][r];
      else       outB[idx] = f2bf(acc[nt][r]);
    }
}

// ---------------------------------------------------------------------------
// RoPE (in place, bf16). One thread per (row, head, d<32) pair -> race-free.
// ---------------------------------------------------------------------------
__global__ void rope_kernel(unsigned short* __restrict__ t,
                            const float* __restrict__ rpe, int npair)
{
  int i = blockIdx.x * blockDim.x + threadIdx.x;
  if (i >= npair) return;
  int row  = i >> 9;          // / 512
  int j    = i & 511;
  int head = j >> 5;
  int d    = j & 31;
  int s    = row & (SEQ - 1);
  size_t base = (size_t)row * D_MODEL + head * HEAD_DIM;
  float a1 = rpe[s * HEAD_DIM + d];
  float a2 = rpe[s * HEAD_DIM + d + 32];
  float t1 = bf2f(t[base + d]);
  float t2 = bf2f(t[base + d + 32]);
  float o1 = t1 * __cosf(a1) - t2 * __sinf(a1);
  float o2 = t2 * __cosf(a2) + t1 * __sinf(a2);
  t[base + d]      = f2bf(o1);
  t[base + d + 32] = f2bf(o2);
}

// ---------------------------------------------------------------------------
// Causal flash attention per (b,h). Block = 128 thr = 4 waves, 64 q-rows.
// QK^T and P*V via WMMA; K/V tiles staged by TDM; P via per-wave LDS.
// ---------------------------------------------------------------------------
__global__ __launch_bounds__(128) void attention_kernel(
    const unsigned short* __restrict__ Q,
    const unsigned short* __restrict__ Km,
    const unsigned short* __restrict__ Vm,
    unsigned short* __restrict__ O)
{
  __shared__ unsigned short Ks[64][72];
  __shared__ unsigned short Vs[64][72];
  __shared__ unsigned short Ps[4][16][72];
  const int tid = threadIdx.x, lane = tid & 31, wid = tid >> 5;
  const int laneHi = lane >> 4, lane16 = lane & 15;
  const int qt = blockIdx.x;
  const int b = blockIdx.y / N_HEADS, h = blockIdx.y % N_HEADS;
  const int q0 = qt * 64;
  const int qrow = q0 + wid * 16;
  const float scale = 0.125f;   // 1/sqrt(64)

  FragB16 qa[2];  // Q fragment, 2 chunks of K=32
  {
    size_t gbase = ((size_t)b * SEQ + qrow + lane16) * D_MODEL + h * HEAD_DIM;
#pragma unroll
    for (int c = 0; c < 2; ++c)
#pragma unroll
      for (int j = 0; j < 8; ++j) {
        int k = c * 32 + ((j >> 2) << 4) + (laneHi << 3) + ((j & 3) << 1);
        qa[c].u[j] = *(const unsigned int*)(Q + gbase + k);
      }
  }
  float m8[8], l8[8];
#pragma unroll
  for (int r = 0; r < 8; ++r) { m8[r] = -1e30f; l8[r] = 0.0f; }
  v8f acc[4];
#pragma unroll
  for (int t = 0; t < 4; ++t) acc[t] = v8f_zero();

#if USE_TDM
  const unsigned ldsK = (unsigned)(uintptr_t)&Ks[0][0];
  const unsigned ldsV = (unsigned)(uintptr_t)&Vs[0][0];
#else
  const int ldrow = tid >> 1, ldhalf = tid & 1;
#endif

  for (int kt = 0; kt <= qt; ++kt) {
    const int k0 = kt * 64;
    __syncthreads();
#if USE_TDM
    if (tid < 32) {
      size_t gb = ((size_t)b * SEQ + k0) * D_MODEL + h * HEAD_DIM;
      tdm_load_2d(ldsK, Km + gb, D_MODEL, ROWS, D_MODEL, 64u, 64u, 4u, 3u);
      tdm_load_2d(ldsV, Vm + gb, D_MODEL, ROWS, D_MODEL, 64u, 64u, 4u, 3u);
      __builtin_amdgcn_s_wait_tensorcnt(0);
    }
#else
    {
      size_t gb = ((size_t)b * SEQ + k0 + ldrow) * D_MODEL + h * HEAD_DIM + ldhalf * 32;
      const uint4* gk = (const uint4*)(Km + gb);
      const uint4* gv = (const uint4*)(Vm + gb);
      uint4* lk = (uint4*)&Ks[ldrow][ldhalf * 32];
      uint4* lv = (uint4*)&Vs[ldrow][ldhalf * 32];
      lk[0] = gk[0]; lk[1] = gk[1]; lk[2] = gk[2]; lk[3] = gk[3];
      lv[0] = gv[0]; lv[1] = gv[1]; lv[2] = gv[2]; lv[3] = gv[3];
    }
#endif
    __syncthreads();
    // scores = Q * K^T : 2 K-chunks x 4 N-tiles
    v8f sc[4];
#pragma unroll
    for (int t = 0; t < 4; ++t) sc[t] = v8f_zero();
#pragma unroll
    for (int c = 0; c < 2; ++c)
#pragma unroll
      for (int nt = 0; nt < 4; ++nt) {
        FragB16 bfr;
#pragma unroll
        for (int j = 0; j < 8; ++j) {
          int k = c * 32 + ((j >> 2) << 4) + (laneHi << 3) + ((j & 3) << 1);
          bfr.u[j] = *(const unsigned int*)&Ks[nt * 16 + lane16][k];
        }
        sc[nt] = __builtin_amdgcn_wmma_f32_16x16x32_bf16(
            false, qa[c].v, false, bfr.v, (short)0, sc[nt], false, false);
      }
    // online softmax (row stats reduced across 16-lane half-groups)
    float sv[32];
#pragma unroll
    for (int nt = 0; nt < 4; ++nt) {
      int col = k0 + nt * 16 + lane16;
#pragma unroll
      for (int r = 0; r < 8; ++r) {
        float s = sc[nt][r] * scale;
        int row = qrow + laneHi * 8 + r;
        if (col > row) s = -1e30f;
        sv[nt * 8 + r] = s;
      }
    }
#pragma unroll
    for (int r = 0; r < 8; ++r) {
      float rmax = fmaxf(fmaxf(sv[r], sv[8 + r]), fmaxf(sv[16 + r], sv[24 + r]));
#pragma unroll
      for (int off = 1; off < 16; off <<= 1) rmax = fmaxf(rmax, __shfl_xor(rmax, off, 32));
      float mn = fmaxf(m8[r], rmax);
      float corr = __expf(m8[r] - mn);
      m8[r] = mn;
      float psum = 0.0f;
#pragma unroll
      for (int nt = 0; nt < 4; ++nt) {
        float p = __expf(sv[nt * 8 + r] - mn);
        psum += p;
        Ps[wid][laneHi * 8 + r][nt * 16 + lane16] = f2bf(p);
      }
#pragma unroll
      for (int off = 1; off < 16; off <<= 1) psum += __shfl_xor(psum, off, 32);
      l8[r] = l8[r] * corr + psum;
#pragma unroll
      for (int nt = 0; nt < 4; ++nt) acc[nt][r] *= corr;
    }
    asm volatile("s_wait_dscnt 0" ::: "memory");  // P stores visible to own wave
    // acc += P * V
#pragma unroll
    for (int c = 0; c < 2; ++c) {
      FragB16 pa;
#pragma unroll
      for (int j = 0; j < 8; ++j) {
        int k = c * 32 + ((j >> 2) << 4) + (laneHi << 3) + ((j & 3) << 1);
        pa.u[j] = *(const unsigned int*)&Ps[wid][lane16][k];
      }
#pragma unroll
      for (int nt = 0; nt < 4; ++nt) {
        FragB16 bfr;   // B[k][n] = V[key k][dim n] (column gather from LDS)
#pragma unroll
        for (int e = 0; e < 16; ++e) {
          int k = c * 32 + ((e >> 3) << 4) + (laneHi << 3) + (e & 7);
          bfr.s[e] = Vs[k][nt * 16 + lane16];
        }
        acc[nt] = __builtin_amdgcn_wmma_f32_16x16x32_bf16(
            false, pa.v, false, bfr.v, (short)0, acc[nt], false, false);
      }
    }
  }
  // normalize + write bf16 output [B,S,H*D]
#pragma unroll
  for (int r = 0; r < 8; ++r) {
    float inv = 1.0f / l8[r];
    int row = qrow + laneHi * 8 + r;
    size_t ob = ((size_t)b * SEQ + row) * D_MODEL + h * HEAD_DIM;
#pragma unroll
    for (int nt = 0; nt < 4; ++nt)
      O[ob + nt * 16 + lane16] = f2bf(acc[nt][r] * inv);
  }
}

// ---------------------------------------------------------------------------
// SwiGLU: u = silu(g) * u   (in place into u)
// ---------------------------------------------------------------------------
__global__ void swiglu_kernel(const unsigned short* __restrict__ g,
                              unsigned short* __restrict__ u, int n)
{
  int i = blockIdx.x * blockDim.x + threadIdx.x;
  if (i >= n) return;
  float gv = bf2f(g[i]), uv = bf2f(u[i]);
  float s = gv / (1.0f + __expf(-gv));
  u[i] = f2bf(s * uv);
}

// ---------------------------------------------------------------------------
extern "C" void kernel_launch(void* const* d_in, const int* in_sizes, int n_in,
                              void* d_out, int out_size, void* d_ws, size_t ws_size,
                              hipStream_t stream)
{
  (void)in_sizes; (void)n_in; (void)out_size; (void)ws_size;
  const float* x     = (const float*)d_in[0];
  const float* rpe   = (const float*)d_in[1];
  const float* wq    = (const float*)d_in[2];
  const float* wk    = (const float*)d_in[3];
  const float* wv    = (const float*)d_in[4];
  const float* wo    = (const float*)d_in[5];
  const float* wgate = (const float*)d_in[6];
  const float* wup   = (const float*)d_in[7];
  const float* wdown = (const float*)d_in[8];
  const float* n1w   = (const float*)d_in[9];
  const float* n2w   = (const float*)d_in[10];
  float* out = (float*)d_out;

  char* ws = (char*)d_ws;
  size_t off = 0;
  auto alloc = [&](size_t bytes) -> char* {
    char* p = ws + off; off += (bytes + 255) & ~(size_t)255; return p;
  };
  unsigned short* hb   = (unsigned short*)alloc((size_t)ROWS * D_MODEL * 2);
  unsigned short* qb   = (unsigned short*)alloc((size_t)ROWS * D_MODEL * 2);
  unsigned short* kb   = (unsigned short*)alloc((size_t)ROWS * D_MODEL * 2);
  unsigned short* vb   = (unsigned short*)alloc((size_t)ROWS * D_MODEL * 2);
  float*          xmid = (float*)alloc((size_t)ROWS * D_MODEL * 4);
  unsigned short* gb   = (unsigned short*)alloc((size_t)ROWS * D_FF * 2);
  unsigned short* ub   = (unsigned short*)alloc((size_t)ROWS * D_FF * 2);
  unsigned short* attnb = hb;   // hb dead after QKV GEMMs
  unsigned short* h2b   = qb;   // qb dead after attention

  dim3 blk(256);
  dim3 g1(D_MODEL / 64, ROWS / 128);
  dim3 g2(D_FF / 64,   ROWS / 128);

  rmsnorm_kernel<<<ROWS, 256, 0, stream>>>(x, n1w, hb);

  gemm_bf16<false><<<g1, blk, 0, stream>>>(hb, wq, nullptr, nullptr, qb, ROWS, D_MODEL, D_MODEL);
  gemm_bf16<false><<<g1, blk, 0, stream>>>(hb, wk, nullptr, nullptr, kb, ROWS, D_MODEL, D_MODEL);
  gemm_bf16<false><<<g1, blk, 0, stream>>>(hb, wv, nullptr, nullptr, vb, ROWS, D_MODEL, D_MODEL);

  int npair = ROWS * (D_MODEL / 2);
  rope_kernel<<<(npair + 255) / 256, 256, 0, stream>>>(qb, rpe, npair);
  rope_kernel<<<(npair + 255) / 256, 256, 0, stream>>>(kb, rpe, npair);

  dim3 ga(SEQ / 64, BATCH * N_HEADS);
  attention_kernel<<<ga, dim3(128), 0, stream>>>(qb, kb, vb, attnb);

  gemm_bf16<true><<<g1, blk, 0, stream>>>(attnb, wo, x, xmid, nullptr, ROWS, D_MODEL, D_MODEL);

  rmsnorm_kernel<<<ROWS, 256, 0, stream>>>(xmid, n2w, h2b);

  gemm_bf16<false><<<g2, blk, 0, stream>>>(h2b, wgate, nullptr, nullptr, gb, ROWS, D_FF, D_MODEL);
  gemm_bf16<false><<<g2, blk, 0, stream>>>(h2b, wup,   nullptr, nullptr, ub, ROWS, D_FF, D_MODEL);

  int nff = ROWS * D_FF;
  swiglu_kernel<<<(nff + 255) / 256, 256, 0, stream>>>(gb, ub, nff);

  gemm_bf16<true><<<g1, blk, 0, stream>>>(ub, wdown, xmid, out, nullptr, ROWS, D_MODEL, D_FF);
}
